// UnrollingModel_51986284150923
// MI455X (gfx1250) — matrix operand order repaired and checked by reading, not compiled, not run on hardware
//
#include <hip/hip_runtime.h>

// ---------------------------------------------------------------------------
// CDNA5 (gfx1250) fused bf16-WMMA GEMM pipeline for the UnrollingModel.
// wave32; D = A x B + C via v_wmma_f32_16x16x32_bf16, fp32 accumulate.
// Block tile: M=128 (8 waves x 16 rows), N=64 (4 accumulators per wave).
// Double-buffered LDS B tile, one barrier per K step.
// ---------------------------------------------------------------------------

typedef __attribute__((ext_vector_type(16))) __bf16 v16bf;
typedef __attribute__((ext_vector_type(8)))  float  v8f;

#define XW    6432   // row stride of x
#define POFF  6240   // pump settings offset in x (PS)
#define KSTEP 32     // K per WMMA step (bf16)
#define NT    64     // N columns per block

// fp32 -> bf16, round-to-nearest-even
__device__ __forceinline__ unsigned short f2bf_u(float f) {
  unsigned u = __builtin_bit_cast(unsigned, f);
  return (unsigned short)((u + 0x7FFFu + ((u >> 16) & 1u)) >> 16);
}
__device__ __forceinline__ __bf16 f2bf(float f) {
  unsigned short s = f2bf_u(f);
  return __builtin_bit_cast(__bf16, s);
}
__device__ __forceinline__ unsigned pack2(float a, float b) {
  return (unsigned)f2bf_u(a) | ((unsigned)f2bf_u(b) << 16);
}
// 8 contiguous fp32 via two b128 loads
__device__ __forceinline__ void load8(const float* __restrict__ p, float* v) {
  float4 x0 = *(const float4*)p;
  float4 x1 = *(const float4*)(p + 4);
  v[0]=x0.x; v[1]=x0.y; v[2]=x0.z; v[3]=x0.w;
  v[4]=x1.x; v[5]=x1.y; v[6]=x1.z; v[7]=x1.w;
}

// ---- guard-free A fragment for a full K step ----
//  AMODE 0: A0[m, k]
//  AMODE 1: q[m,k] * resSq[m,k]      (pump columns live only in the tail)
//  AMODE 2: A0[m,k] * (A1[m,k] + A2[m,k] + A3[m,k])
template<int AMODE>
__device__ __forceinline__ v16bf build_afrag_full(
    const float* __restrict__ A0, const float* __restrict__ A1,
    const float* __restrict__ A2, const float* __restrict__ A3,
    int lda, int m, int k0, int ksub)
{
  float v[16];
  if constexpr (AMODE == 0) {
    const float* p = A0 + (size_t)m * lda + k0 + ksub;
    load8(p, v); load8(p + 16, v + 8);
  } else if constexpr (AMODE == 1) {
    const size_t o = (size_t)m * 4096 + k0 + ksub;
    float a1[16], a2[16];
    load8(A1 + o, a1); load8(A1 + o + 16, a1 + 8);
    load8(A2 + o, a2); load8(A2 + o + 16, a2 + 8);
#pragma unroll
    for (int i = 0; i < 16; ++i) v[i] = a1[i] * a2[i];
  } else {
    const size_t o = (size_t)m * lda + k0 + ksub;
    float a0[16], a1[16], a2[16], a3[16];
    load8(A0 + o, a0); load8(A0 + o + 16, a0 + 8);
    load8(A1 + o, a1); load8(A1 + o + 16, a1 + 8);
    load8(A2 + o, a2); load8(A2 + o + 16, a2 + 8);
    load8(A3 + o, a3); load8(A3 + o + 16, a3 + 8);
#pragma unroll
    for (int i = 0; i < 16; ++i) v[i] = a0[i] * (a1[i] + a2[i] + a3[i]);
  }
  v16bf a;
#pragma unroll
  for (int i = 0; i < 16; ++i) a[i] = f2bf(v[i]);
  return a;
}

// Scalar guarded element for the K tail (only AMODE 1, K=4104 uses it).
template<int AMODE>
__device__ __forceinline__ float a_tail(
    const float* __restrict__ A0, const float* __restrict__ A1,
    const float* __restrict__ A2, const float* __restrict__ A3,
    const float* __restrict__ pump, int lda, int K, int m, int k)
{
  if (k >= K) return 0.0f;
  if constexpr (AMODE == 0) return A0[(size_t)m * lda + k];
  if constexpr (AMODE == 1) {
    if (k < 4096) { size_t o = (size_t)m * 4096 + k; return A1[o] * A2[o]; }
    return pump[(size_t)m * XW + (k - 4096)];
  }
  size_t o = (size_t)m * lda + k;
  return A0[o] * (A1[o] + A2[o] + A3[o]);
}

// C[M=256, N] = act( A[M,K] @ W[N,K]^T + b )
//  EMODE 0: none   EMODE 1: relu   EMODE 2: t=prelu(acc+b); out = qin - t
template<int AMODE, int EMODE, bool TAIL>
__global__ __launch_bounds__(256) void gemm_wmma(
    const float* __restrict__ W, const float* __restrict__ bias,
    const float* __restrict__ A0, const float* __restrict__ A1,
    const float* __restrict__ A2, const float* __restrict__ A3,
    const float* __restrict__ pump, const float* __restrict__ qin,
    const float* __restrict__ prelu, int pidx,
    float* __restrict__ out, int N, int K, int lda)
{
  // double-buffered B tile: [buf][col][k] bf16, 2 x 4KB
  __shared__ __align__(32) unsigned short ldsB[2][NT * KSTEP];

  const int tid  = threadIdx.x;
  const int lane = tid & 31;
  const int wave = tid >> 5;
  const int n0   = blockIdx.x * NT;
  const int mblk = blockIdx.y * 128;

  // 16-bit A 16x32 lane layout: lanes 0-15: K {0..7,16..23}; 16-31: {8..15,24..31}
  const int mrow = mblk + wave * 16 + (lane & 15);
  const int khal = lane >> 4;
  const int ksub = khal * 8;
  const int loff = (lane & 15) * KSTEP + khal * 16;   // B-frag base (ushorts)

  // staging: 256 threads cover 64 W rows x 32 k, 8 fp32 each
  const int srow = tid >> 2;          // 0..63
  const int skq  = (tid & 3) * 8;     // 0,8,16,24
  const size_t wrow = (size_t)(n0 + srow) * K;
  const int sdst = srow * KSTEP + (tid & 1) * 16 + ((tid >> 1) & 1) * 8;

  v8f acc[4];
#pragma unroll
  for (int i = 0; i < 4; ++i)
    acc[i] = (v8f){0.f, 0.f, 0.f, 0.f, 0.f, 0.f, 0.f, 0.f};

  const int kfull = K / KSTEP;

  // prologue: stage tile 0 into buffer 0
  {
    const float* wp = W + wrow + skq;
    float4 w0 = *(const float4*)wp;
    float4 w1 = *(const float4*)(wp + 4);
    uint4 pk;
    pk.x = pack2(w0.x, w0.y); pk.y = pack2(w0.z, w0.w);
    pk.z = pack2(w1.x, w1.y); pk.w = pack2(w1.z, w1.w);
    *(uint4*)&ldsB[0][sdst] = pk;
  }

  for (int ks = 0; ks < kfull; ++ks) {
    const int k0  = ks * KSTEP;
    const int cur = ks & 1;
    __syncthreads();   // tile `cur` visible; buffer `cur^1` free to overwrite

    // issue global loads for the next W tile (overlaps the WMMA chain)
    const bool have_next = (ks + 1 < kfull);
    float4 w0 = {0.f,0.f,0.f,0.f}, w1 = {0.f,0.f,0.f,0.f};
    if (have_next) {
      const float* wp = W + wrow + (k0 + KSTEP) + skq;
      w0 = *(const float4*)wp;
      w1 = *(const float4*)(wp + 4);
      __builtin_prefetch(wp + KSTEP, 0, 3);   // K tile ks+2
    }

    // A fragment (fused elementwise ops, fp32 -> bf16)
    const v16bf afrag = build_afrag_full<AMODE>(A0, A1, A2, A3, lda, mrow, k0, ksub);

    // all four B fragments first, then the WMMA chain
    const unsigned short* bb = &ldsB[cur][loff];
    v16bf b0 = *(const v16bf*)(bb + 0 * 16 * KSTEP);
    v16bf b1 = *(const v16bf*)(bb + 1 * 16 * KSTEP);
    v16bf b2 = *(const v16bf*)(bb + 2 * 16 * KSTEP);
    v16bf b3 = *(const v16bf*)(bb + 3 * 16 * KSTEP);
    acc[0] = __builtin_amdgcn_wmma_f32_16x16x32_bf16(false, afrag, false, b0, (short)0, acc[0], false, false);
    acc[1] = __builtin_amdgcn_wmma_f32_16x16x32_bf16(false, afrag, false, b1, (short)0, acc[1], false, false);
    acc[2] = __builtin_amdgcn_wmma_f32_16x16x32_bf16(false, afrag, false, b2, (short)0, acc[2], false, false);
    acc[3] = __builtin_amdgcn_wmma_f32_16x16x32_bf16(false, afrag, false, b3, (short)0, acc[3], false, false);

    // stage next tile into the other buffer (readers sync at next barrier)
    if (have_next) {
      uint4 pk;
      pk.x = pack2(w0.x, w0.y); pk.y = pack2(w0.z, w0.w);
      pk.z = pack2(w1.x, w1.y); pk.w = pack2(w1.z, w1.w);
      *(uint4*)&ldsB[cur ^ 1][sdst] = pk;
    }
  }

  if constexpr (TAIL) {                       // one guarded partial K step
    const int k0 = kfull * KSTEP;
    __syncthreads();
    {
      unsigned short* dst = &ldsB[0][sdst];
#pragma unroll
      for (int j = 0; j < 8; ++j) {
        float v = (k0 + skq + j < K) ? W[wrow + k0 + skq + j] : 0.f;
        dst[j] = f2bf_u(v);
      }
    }
    __syncthreads();
    v16bf afrag;
#pragma unroll
    for (int i = 0; i < 16; ++i) {
      int k = k0 + (i < 8 ? (ksub + i) : (16 + ksub + (i - 8)));
      afrag[i] = f2bf(a_tail<AMODE>(A0, A1, A2, A3, pump, lda, K, mrow, k));
    }
    const unsigned short* bb = &ldsB[0][loff];
    v16bf b0 = *(const v16bf*)(bb + 0 * 16 * KSTEP);
    v16bf b1 = *(const v16bf*)(bb + 1 * 16 * KSTEP);
    v16bf b2 = *(const v16bf*)(bb + 2 * 16 * KSTEP);
    v16bf b3 = *(const v16bf*)(bb + 3 * 16 * KSTEP);
    acc[0] = __builtin_amdgcn_wmma_f32_16x16x32_bf16(false, afrag, false, b0, (short)0, acc[0], false, false);
    acc[1] = __builtin_amdgcn_wmma_f32_16x16x32_bf16(false, afrag, false, b1, (short)0, acc[1], false, false);
    acc[2] = __builtin_amdgcn_wmma_f32_16x16x32_bf16(false, afrag, false, b2, (short)0, acc[2], false, false);
    acc[3] = __builtin_amdgcn_wmma_f32_16x16x32_bf16(false, afrag, false, b3, (short)0, acc[3], false, false);
  }

  // ---- epilogue: C layout lanes 0-15 -> M=r, lanes 16-31 -> M=8+r ----
  const int mbase = mblk + wave * 16 + khal * 8;
  const int cloc  = lane & 15;
  float pa = 0.f;
  if constexpr (EMODE == 2) pa = prelu[pidx];
#pragma unroll
  for (int ni = 0; ni < 4; ++ni) {
    const int   col = n0 + ni * 16 + cloc;
    const float bv  = bias[col];
#pragma unroll
    for (int r = 0; r < 8; ++r) {
      const size_t o = (size_t)(mbase + r) * N + col;
      float v = acc[ni][r] + bv;
      if constexpr (EMODE == 1) v = fmaxf(v, 0.f);
      if constexpr (EMODE == 2) { v = (v >= 0.f) ? v : pa * v; v = qin[o] - v; }
      out[o] = v;
    }
  }
}

// q = (pi/4) * d * d,  d = x[:, 96:96+4096]
__global__ __launch_bounds__(256) void qinit_kernel(
    const float* __restrict__ x, float* __restrict__ q) {
  int idx = blockIdx.x * blockDim.x + threadIdx.x;   // 256*4096 total
  int m = idx >> 12, j = idx & 4095;
  float d = x[(size_t)m * XW + 96 + j];
  q[idx] = 0.78539816339744830962f * d * d;
}

template<int AMODE, int EMODE, bool TAIL>
static inline void launch_gemm(hipStream_t s,
    const float* W, const float* bias,
    const float* A0, const float* A1, const float* A2, const float* A3,
    const float* pump, const float* qin, const float* prelu, int pidx,
    float* out, int N, int K, int lda)
{
  dim3 grid(N / NT, 2);
  gemm_wmma<AMODE, EMODE, TAIL><<<grid, dim3(256), 0, s>>>(
      W, bias, A0, A1, A2, A3, pump, qin, prelu, pidx, out, N, K, lda);
}

extern "C" void kernel_launch(void* const* d_in, const int* in_sizes, int n_in,
                              void* d_out, int out_size, void* d_ws, size_t ws_size,
                              hipStream_t stream) {
  (void)in_sizes; (void)n_in; (void)out_size; (void)ws_size;

  const float* x      = (const float*)d_in[0];
  const float* W_h0_q = (const float*)d_in[2];  const float* b_h0_q = (const float*)d_in[3];
  const float* W_h0_h = (const float*)d_in[4];  const float* b_h0_h = (const float*)d_in[5];
  const float* W_S    = (const float*)d_in[6];  const float* b_S    = (const float*)d_in[7];
  const float* W_q0_h = (const float*)d_in[8];  const float* b_q0_h = (const float*)d_in[9];
  const float* W_s_q  = (const float*)d_in[10]; const float* b_s_q  = (const float*)d_in[11];
  const float* W_out  = (const float*)d_in[12]; const float* b_out  = (const float*)d_in[13];
  const float* W_hf   = (const float*)d_in[14]; const float* b_hf   = (const float*)d_in[15];
  const float* W_fh   = (const float*)d_in[16]; const float* b_fh   = (const float*)d_in[17];
  const float* W_resq = (const float*)d_in[18]; const float* b_resq = (const float*)d_in[19];
  const float* W_Dq   = (const float*)d_in[20]; const float* b_Dq   = (const float*)d_in[21];
  const float* W_Dh   = (const float*)d_in[22]; const float* b_Dh   = (const float*)d_in[23];
  const float* prelu  = (const float*)d_in[24];

  const int F = 4096, H = 2112, JD = 2048, BH = 96;
  float* ws = (float*)d_ws;
  const size_t SZF = (size_t)256 * F, SZH = (size_t)256 * H;
  float* q    = ws;            // (256,4096)  running flow state
  float* rh0q = q    + SZF;    // res_h0_q
  float* rSq  = rh0q + SZF;    // res_S_q
  float* rsq  = rSq  + SZF;    // res_s_q
  float* Dq   = rsq  + SZF;    // D_q
  float* rh0h = Dq   + SZF;    // res_h0_h (256,2112)
  float* rqh  = rh0h + SZH;    // res_q_h
  float* Dh   = rqh  + SZH;    // D_h
  float* hbuf = Dh   + SZH;    // h

  // q = pi/4 * d^2
  qinit_kernel<<<dim3((256 * F) / 256), dim3(256), 0, stream>>>(x, q);

  // Pre-loop residual streams (A slices read strided straight out of x).
  launch_gemm<0,0,false>(stream, W_h0_q, b_h0_q, x,        nullptr, nullptr, nullptr, nullptr, nullptr, nullptr, 0, rh0q, F, BH, XW);
  launch_gemm<0,0,false>(stream, W_h0_h, b_h0_h, x,        nullptr, nullptr, nullptr, nullptr, nullptr, nullptr, 0, rh0h, H, BH, XW);
  launch_gemm<0,1,false>(stream, W_S,    b_S,    x + 96,   nullptr, nullptr, nullptr, nullptr, nullptr, nullptr, 0, rSq,  F, F,  XW);
  launch_gemm<0,0,false>(stream, W_q0_h, b_q0_h, q,        nullptr, nullptr, nullptr, nullptr, nullptr, nullptr, 0, rqh,  H, F,  F);
  launch_gemm<0,0,false>(stream, W_s_q,  b_s_q,  x + 4192, nullptr, nullptr, nullptr, nullptr, nullptr, nullptr, 0, rsq,  F, JD, XW);

  for (int i = 0; i < 6; ++i) {
    // D_q = Lin(concat(q*res_S_q, pump)); K = 4104 (tail step carries pump)
    launch_gemm<1,0,true>(stream, W_Dq + (size_t)i * F * (F + 8), b_Dq + (size_t)i * F,
                          nullptr, q, rSq, nullptr, x + POFF, nullptr, nullptr, 0,
                          Dq, F, F + 8, F);
    // D_h = relu(Lin(D_q))
    launch_gemm<0,1,false>(stream, W_Dh + (size_t)i * H * F, b_Dh + (size_t)i * H,
                           Dq, nullptr, nullptr, nullptr, nullptr, nullptr, nullptr, 0,
                           Dh, H, F, F);
    // h = relu(Lin(D_q * (q + res_s_q + res_h0_q)))
    launch_gemm<2,1,false>(stream, W_fh + (size_t)i * H * F, b_fh + (size_t)i * H,
                           Dq, q, rsq, rh0q, nullptr, nullptr, nullptr, 0,
                           hbuf, H, F, F);
    // t = Lin((h + res_h0_h + res_q_h) * D_h);  q = q - prelu(t)  (in place)
    launch_gemm<2,2,false>(stream, W_hf + (size_t)i * F * H, b_hf + (size_t)i * F,
                           Dh, hbuf, rh0h, rqh, nullptr, q, prelu, i,
                           q, F, H, H);
    // res_q_h = relu(Lin(q))
    launch_gemm<0,1,false>(stream, W_resq + (size_t)i * H * F, b_resq + (size_t)i * H,
                           q, nullptr, nullptr, nullptr, nullptr, nullptr, nullptr, 0,
                           rqh, H, F, F);
  }

  // out = Lin(q, W_out)
  launch_gemm<0,0,false>(stream, W_out, b_out,
                         q, nullptr, nullptr, nullptr, nullptr, nullptr, nullptr, 0,
                         (float*)d_out, H, F, F);
}
